// MPNN_74466142978399
// MI455X (gfx1250) — matrix operand order repaired and checked by reading, not compiled, not run on hardware
//
#include <hip/hip_runtime.h>
#include <hip/hip_bf16.h>

typedef __attribute__((ext_vector_type(16))) __bf16 bf16x16;
typedef __attribute__((ext_vector_type(8)))  float  f32x8;

// ---------------------------------------------------------------------------
// Fully-static gather-GEMM:
//   out[r, 0:NT*16] = act( concat(seg0..seg3)[r, 0:K] @ W + b )
// Template params: NT = number of 16-wide N tiles, W0..W3 = segment widths
// (0 => segment absent). K/Kpad/segment map are compile-time; the K loop is
// fully unrolled so each instantiation emits Kpad/32 * NT back-to-back
// v_wmma_f32_16x16x32_bf16 with fixed accumulator tuples.
// ---------------------------------------------------------------------------
template<int NT, int W0, int W1, int W2, int W3>
__global__ __launch_bounds__(128) void gemm_gather_kernel(
    const float* __restrict__ b0, const int* __restrict__ i0,
    const float* __restrict__ b1, const int* __restrict__ i1,
    const float* __restrict__ b2, const int* __restrict__ i2,
    const float* __restrict__ b3, const int* __restrict__ i3,
    const __bf16* __restrict__ WT, const float* __restrict__ bias,
    float* __restrict__ out, const int* __restrict__ scat,
    int R, int relu)
{
    constexpr int K    = W0 + W1 + W2 + W3;
    constexpr int Kpad = (K + 31) & ~31;
    constexpr int DOUT = NT * 16;

    __shared__ __bf16 lwt[DOUT * Kpad];           // transposed bf16 weights
    {
        constexpr int TOTAL_DW = (DOUT * Kpad) / 2;
        const uint32_t* s = (const uint32_t*)WT;
        uint32_t* d = (uint32_t*)lwt;
        for (int t = threadIdx.x; t < TOTAL_DW; t += blockDim.x) d[t] = s[t];
    }
    __syncthreads();

    const int lane = threadIdx.x & 31;
    const int wave = threadIdx.x >> 5;
    const int r0 = (blockIdx.x * 4 + wave) * 16;  // 16-row tile per wave
    if (r0 >= R) return;                          // uniform per wave
    const int hlf = lane >> 4;
    const int m   = lane & 15;
    const int rA  = r0 + m;
    const int rr  = rA < R ? rA : R - 1;          // clamp; stores guarded later

    // Per-lane gathered row pointers (compile-time dead if width==0).
    const float* p0 = b0 + (size_t)(i0 ? i0[rr] : rr) * W0;
    const float* p1 = b1 + (size_t)(i1 ? i1[rr] : rr) * W1;
    const float* p2 = nullptr;
    const float* p3 = nullptr;
    if constexpr (W2 > 0) p2 = b2 + (size_t)(i2 ? i2[rr] : rr) * W2;
    if constexpr (W3 > 0) p3 = b3 + (size_t)(i3 ? i3[rr] : rr) * W3;

    f32x8 acc[NT];
    #pragma unroll
    for (int t = 0; t < NT; ++t)
        #pragma unroll
        for (int j = 0; j < 8; ++j) acc[t][j] = 0.f;

    #pragma unroll
    for (int kt = 0; kt < Kpad; kt += 32) {
        // A fragment (16x32 bf16): lanes<16 hold K kt..kt+15 contiguous;
        // lanes>=16 hold kt+16..kt+31. Each 8-chunk lies in one segment
        // since widths are 16-multiples.
        bf16x16 a;
        #pragma unroll
        for (int c = 0; c < 2; ++c) {
            const int kc = kt + hlf * 16 + c * 8;
            const int di = c * 8;
            if (kc >= K) {                        // compile-time padded tail
                #pragma unroll
                for (int j = 0; j < 8; ++j) a[di + j] = (__bf16)0.f;
            } else {
                const float* p;
                if      (kc < W0)           p = p0 + kc;
                else if (kc < W0 + W1)      p = p1 + (kc - W0);
                else if (kc < W0 + W1 + W2) p = p2 + (kc - W0 - W1);
                else                        p = p3 + (kc - W0 - W1 - W2);

                const float4 f0 = *(const float4*)(p);
                const float4 f1 = *(const float4*)(p + 4);
                a[di + 0] = (__bf16)f0.x; a[di + 1] = (__bf16)f0.y;
                a[di + 2] = (__bf16)f0.z; a[di + 3] = (__bf16)f0.w;
                a[di + 4] = (__bf16)f1.x; a[di + 5] = (__bf16)f1.y;
                a[di + 6] = (__bf16)f1.z; a[di + 7] = (__bf16)f1.w;
            }
        }
        // B fragments from LDS (WT stored [n][Kpad]): lane n<16 -> col n,
        // K kt..kt+15 contiguous; lanes>=16 -> col n-16, K kt+16..kt+31.
        const int ks = kt + hlf * 16;
        #pragma unroll
        for (int nt = 0; nt < NT; ++nt) {
            const bf16x16 bm = *(const bf16x16*)&lwt[(nt * 16 + m) * Kpad + ks];
            acc[nt] = __builtin_amdgcn_wmma_f32_16x16x32_bf16(
                false, a, false, bm, (short)0, acc[nt], false, false);
        }
    }

    // Epilogue: C layout: lane -> col N = m (+16*nt), VGPR j -> row M = j+8*hlf.
    #pragma unroll
    for (int nt = 0; nt < NT; ++nt) {
        #pragma unroll
        for (int j = 0; j < 8; ++j) {
            const int row = r0 + j + hlf * 8;
            if (row < R) {
                const int n = nt * 16 + m;
                float v = acc[nt][j] + bias[n];
                if (relu) v = fmaxf(v, 0.f);
                if (scat) atomicAdd(&out[(size_t)scat[row] * DOUT + n], v);
                else      out[(size_t)row * DOUT + n] = v;
            }
        }
    }
}

// ---------------------------------------------------------------------------
// Support kernels
// ---------------------------------------------------------------------------
__global__ void conv_weight_kernel(const float* __restrict__ W,
                                   __bf16* __restrict__ WT, int K, int dout)
{
    const int Kpad = (K + 31) & ~31;
    const int i = blockIdx.x * blockDim.x + threadIdx.x;
    if (i < dout * Kpad) {
        const int n = i / Kpad, k = i % Kpad;
        WT[(size_t)n * Kpad + k] = (k < K) ? (__bf16)W[(size_t)k * dout + n]
                                           : (__bf16)0.f;
    }
}

__global__ void compose_idx_kernel(const int* __restrict__ tab,
                                   const int* __restrict__ idx,
                                   int* __restrict__ out, int n)
{
    const int i = blockIdx.x * blockDim.x + threadIdx.x;
    if (i < n) out[i] = tab[idx[i]];
}

__global__ void zero_f_kernel(float* __restrict__ p, long long n)
{
    long long i = (long long)blockIdx.x * blockDim.x + threadIdx.x;
    const long long stride = (long long)gridDim.x * blockDim.x;
    for (; i < n; i += stride) p[i] = 0.f;
}

__global__ void scatter_sum_kernel(const float* __restrict__ in,
                                   const int* __restrict__ idx,
                                   float* __restrict__ out, int R, int F)
{
    const long long i = (long long)blockIdx.x * blockDim.x + threadIdx.x;
    if (i < (long long)R * F) {
        const int r = (int)(i / F), f = (int)(i % F);
        atomicAdd(&out[(size_t)idx[r] * F + f], in[i]);
    }
}

// per-feature sum & sum-of-squares; LDS (ds_add_f32) block reduction
__global__ void bn_stats_kernel(const float* __restrict__ z,
                                float* __restrict__ stats, int R, int F)
{
    __shared__ float ls[128];
    for (int t = threadIdx.x; t < 2 * F; t += blockDim.x) ls[t] = 0.f;
    __syncthreads();
    const int r = blockIdx.x * blockDim.x + threadIdx.x;
    if (r < R) {
        const float* p = z + (size_t)r * F;
        for (int f = 0; f < F; ++f) {
            const float v = p[f];
            atomicAdd(&ls[f], v);
            atomicAdd(&ls[F + f], v * v);
        }
    }
    __syncthreads();
    for (int t = threadIdx.x; t < 2 * F; t += blockDim.x) atomicAdd(&stats[t], ls[t]);
}

__global__ void bn_apply_kernel(float* __restrict__ z,
                                const float* __restrict__ stats,
                                const float* __restrict__ g,
                                const float* __restrict__ b, int R, int F)
{
    const long long i = (long long)blockIdx.x * blockDim.x + threadIdx.x;
    if (i < (long long)R * F) {
        const int f = (int)(i % F);
        const float mu  = stats[f] / (float)R;
        const float var = stats[F + f] / (float)R - mu * mu;
        z[i] = (z[i] - mu) * rsqrtf(var + 1e-5f) * g[f] + b[f];
    }
}

// ---------------------------------------------------------------------------
// Host helpers
// ---------------------------------------------------------------------------
static void zero_launch(float* p, long long n, hipStream_t stream)
{
    int grid = (int)((n + 255) / 256); if (grid > 4096) grid = 4096;
    hipLaunchKernelGGL(zero_f_kernel, dim3(grid), dim3(256), 0, stream, p, n);
}

template<int NT, int W0, int W1, int W2, int W3>
static void launch_gemm(const float* s0, const int* j0, const float* s1, const int* j1,
                        const float* s2, const int* j2, const float* s3, const int* j3,
                        const __bf16* WT, const float* bias, float* out,
                        const int* scat, int R, int relu, hipStream_t stream)
{
    hipLaunchKernelGGL((gemm_gather_kernel<NT, W0, W1, W2, W3>),
                       dim3((R + 63) / 64), dim3(128), 0, stream,
                       s0, j0, s1, j1, s2, j2, s3, j3, WT, bias, out, scat, R, relu);
}

// One MetaLayer step (edge MLP, node MLP1+scatter, node MLP2, global MLP).
template<int ED, int UD, int GNT>
static void run_layer(const float* xcur, const float* ecur, const float* ucur,
                      float* xnext, float* enext, float* unext,
                      float* agg, float* nagg, float* eagg,
                      const int* rowi, const int* coli,
                      const int* brow, const int* bcol, const int* batch,
                      const __bf16* wtE, const __bf16* wtN1,
                      const __bf16* wtN2, const __bf16* wtG,
                      const float* eB, const float* n1B,
                      const float* n2B, const float* gB,
                      int relu, hipStream_t stream)
{
    const int NN = 50000, NE = 800000, NG = 64;

    // EdgeModel: concat(x[col], x[row], e, u[batch[row]])
    launch_gemm<4, 64, 64, ED, UD>(xcur, coli, xcur, rowi, ecur, nullptr,
                                   ucur, brow, wtE, eB, enext, nullptr,
                                   NE, relu, stream);
    // NodeModel MLP1 + fused scatter-sum over col
    zero_launch(agg, (long long)NN * 64, stream);
    launch_gemm<4, 64, 64, 64, UD>(xcur, coli, xcur, rowi, enext, nullptr,
                                   ucur, bcol, wtN1, n1B, agg, coli,
                                   NE, relu, stream);
    // NodeModel MLP2: concat(x, agg, u[batch])
    launch_gemm<4, 64, 64, UD, 0>(xcur, nullptr, agg, nullptr, ucur, batch,
                                  nullptr, nullptr, wtN2, n2B, xnext, nullptr,
                                  NN, relu, stream);
    // GlobalModel: segment sums then small GEMM over 64 graphs
    zero_launch(nagg, (long long)NG * 64, stream);
    zero_launch(eagg, (long long)NG * 64, stream);
    hipLaunchKernelGGL(scatter_sum_kernel, dim3((NN * 64 + 255) / 256), dim3(256),
                       0, stream, xnext, batch, nagg, NN, 64);
    hipLaunchKernelGGL(scatter_sum_kernel,
                       dim3((int)(((long long)NE * 64 + 255) / 256)), dim3(256),
                       0, stream, enext, bcol, eagg, NE, 64);
    launch_gemm<GNT, 64, 64, UD, 0>(nagg, nullptr, eagg, nullptr, ucur, nullptr,
                                    nullptr, nullptr, wtG, gB, unext, nullptr,
                                    NG, relu, stream);
}

// ---------------------------------------------------------------------------
// Entry point
// ---------------------------------------------------------------------------
extern "C" void kernel_launch(void* const* d_in, const int* in_sizes, int n_in,
                              void* d_out, int out_size, void* d_ws, size_t ws_size,
                              hipStream_t stream)
{
    (void)in_sizes; (void)out_size; (void)ws_size;
    if (n_in < 55) return;

    const int NN = 50000, NE = 800000, NG = 64;

    const float* x_in = (const float*)d_in[0];
    const float* e_in = (const float*)d_in[1];
    const float* u_in = (const float*)d_in[2];
    const int*   eidx = (const int*)d_in[53];
    const int*   batch = (const int*)d_in[54];
    const int*   rowi = eidx;        // edge_index[0] = src
    const int*   coli = eidx + NE;   // edge_index[1] = dest

    // jax pytree flatten order (list order, dict keys sorted):
    // bn layers: [bn.edge.b, bn.edge.g, bn.glob.b, bn.glob.g, bn.node.b,
    //             bn.node.g, edge.W, edge.b, glob.W, glob.b, n1.W, n1.b,
    //             n2.W, n2.b]; last layer drops the 6 bn arrays.
    const int pbase[4] = {3, 17, 31, 45};
    const float *eW[4], *eB[4], *gW[4], *gB[4], *n1W[4], *n1B[4], *n2W[4], *n2B[4];
    const float *bnEb[3], *bnEg[3], *bnGb[3], *bnGg[3], *bnNb[3], *bnNg[3];
    for (int L = 0; L < 4; ++L) {
        const int p = pbase[L];
        if (L < 3) {
            bnEb[L] = (const float*)d_in[p + 0]; bnEg[L] = (const float*)d_in[p + 1];
            bnGb[L] = (const float*)d_in[p + 2]; bnGg[L] = (const float*)d_in[p + 3];
            bnNb[L] = (const float*)d_in[p + 4]; bnNg[L] = (const float*)d_in[p + 5];
            eW[L]  = (const float*)d_in[p + 6];  eB[L]  = (const float*)d_in[p + 7];
            gW[L]  = (const float*)d_in[p + 8];  gB[L]  = (const float*)d_in[p + 9];
            n1W[L] = (const float*)d_in[p + 10]; n1B[L] = (const float*)d_in[p + 11];
            n2W[L] = (const float*)d_in[p + 12]; n2B[L] = (const float*)d_in[p + 13];
        } else {
            eW[L]  = (const float*)d_in[p + 0];  eB[L]  = (const float*)d_in[p + 1];
            gW[L]  = (const float*)d_in[p + 2];  gB[L]  = (const float*)d_in[p + 3];
            n1W[L] = (const float*)d_in[p + 4];  n1B[L] = (const float*)d_in[p + 5];
            n2W[L] = (const float*)d_in[p + 6];  n2B[L] = (const float*)d_in[p + 7];
        }
    }

    // workspace carve
    char* ws = (char*)d_ws;
    size_t off = 0;
    auto alloc = [&](size_t bytes) -> void* {
        void* p = ws + off;
        off += (bytes + 255) & ~(size_t)255;
        return p;
    };
    float* xb   = (float*)alloc((size_t)NN * 64 * 4);
    float* ebuf = (float*)alloc((size_t)NE * 64 * 4);
    float* ub0  = (float*)alloc((size_t)NG * 64 * 4);
    float* ub1  = (float*)alloc((size_t)NG * 64 * 4);
    float* agg  = (float*)alloc((size_t)NN * 64 * 4);
    int*   brow = (int*)alloc((size_t)NE * 4);
    int*   bcol = (int*)alloc((size_t)NE * 4);
    float* nagg = (float*)alloc((size_t)NG * 64 * 4);
    float* eagg = (float*)alloc((size_t)NG * 64 * 4);
    float* stats = (float*)alloc(2 * 64 * 4);
    __bf16* wt[16];
    for (int i = 0; i < 16; ++i) wt[i] = (__bf16*)alloc((size_t)64 * 256 * 2);

    // d_out regions double as one side of the ping-pong (fully overwritten)
    float* x_out = (float*)d_out;
    float* e_out = x_out + (size_t)NN * 64;
    float* u_out = e_out + (size_t)NE * 64;

    auto conv = [&](const float* W, __bf16* WTp, int K, int dout) {
        const int Kpad = (K + 31) & ~31;
        const int tot = dout * Kpad;
        hipLaunchKernelGGL(conv_weight_kernel, dim3((tot + 255) / 256), dim3(256),
                           0, stream, W, WTp, K, dout);
    };
    auto bn = [&](float* z, int R, const float* g, const float* b) {
        zero_launch(stats, 128, stream);
        hipLaunchKernelGGL(bn_stats_kernel, dim3((R + 255) / 256), dim3(256),
                           0, stream, z, stats, R, 64);
        const long long tot = (long long)R * 64;
        hipLaunchKernelGGL(bn_apply_kernel, dim3((int)((tot + 255) / 256)), dim3(256),
                           0, stream, z, stats, g, b, R, 64);
    };

    // index composition: batch[row], batch[col]
    hipLaunchKernelGGL(compose_idx_kernel, dim3((NE + 255) / 256), dim3(256), 0, stream,
                       batch, rowi, brow, NE);
    hipLaunchKernelGGL(compose_idx_kernel, dim3((NE + 255) / 256), dim3(256), 0, stream,
                       batch, coli, bcol, NE);

    // weight prep (bf16, transposed [n][Kpad], K zero-padded to 32)
    const int ed_in[4] = {32, 64, 64, 64};
    const int ud_in[4] = {16, 64, 64, 64};
    const int go[4]    = {64, 64, 64, 16};
    for (int L = 0; L < 4; ++L) {
        conv(eW[L],  wt[L * 4 + 0], 128 + ed_in[L] + ud_in[L], 64);
        conv(n1W[L], wt[L * 4 + 1], 192 + ud_in[L], 64);
        conv(n2W[L], wt[L * 4 + 2], 128 + ud_in[L], 64);
        conv(gW[L],  wt[L * 4 + 3], 128 + ud_in[L], go[L]);
    }

    // buffer ping-pong (layer 3 lands directly in d_out)
    const float* xcur[4]  = {x_in, xb, x_out, xb};
    float*       xnext[4] = {xb, x_out, xb, x_out};
    const float* ecur[4]  = {e_in, ebuf, e_out, ebuf};
    float*       enext[4] = {ebuf, e_out, ebuf, e_out};
    const float* ucur[4]  = {u_in, ub0, ub1, ub0};
    float*       unext[4] = {ub0, ub1, ub0, u_out};

    for (int L = 0; L < 4; ++L) {
        const int relu = (L < 3) ? 1 : 0;
        if (L == 0) {
            run_layer<32, 16, 4>(xcur[L], ecur[L], ucur[L], xnext[L], enext[L], unext[L],
                                 agg, nagg, eagg, rowi, coli, brow, bcol, batch,
                                 wt[0], wt[1], wt[2], wt[3],
                                 eB[L], n1B[L], n2B[L], gB[L], relu, stream);
        } else if (L < 3) {
            run_layer<64, 64, 4>(xcur[L], ecur[L], ucur[L], xnext[L], enext[L], unext[L],
                                 agg, nagg, eagg, rowi, coli, brow, bcol, batch,
                                 wt[L * 4 + 0], wt[L * 4 + 1], wt[L * 4 + 2], wt[L * 4 + 3],
                                 eB[L], n1B[L], n2B[L], gB[L], relu, stream);
        } else {
            run_layer<64, 64, 1>(xcur[L], ecur[L], ucur[L], xnext[L], enext[L], unext[L],
                                 agg, nagg, eagg, rowi, coli, brow, bcol, batch,
                                 wt[L * 4 + 0], wt[L * 4 + 1], wt[L * 4 + 2], wt[L * 4 + 3],
                                 eB[L], n1B[L], n2B[L], gB[L], relu, stream);
        }
        if (L < 3) {
            bn(xnext[L], NN, bnNg[L], bnNb[L]);
            bn(enext[L], NE, bnEg[L], bnEb[L]);
            bn(unext[L], NG, bnGg[L], bnGb[L]);
        }
    }
}
